// TinyRGNN_49331994362499
// MI455X (gfx1250) — matrix operand order repaired and checked by reading, not compiled, not run on hardware
//
#include <hip/hip_runtime.h>
#include <hip/hip_bf16.h>

// ---------------------------------------------------------------------------
// TinyRGNN on MI455X (gfx1250, wave32, WMMA).
// h (N x 128) kept in f16; segment_sum done with fp32 global atomics;
// all (N x 128) @ (128 x 128) GEMMs use v_wmma_f32_16x16x32_f16.
// Workspace use: ~131 MB.
// ---------------------------------------------------------------------------

#define N_NODES 100000
#define N_REL   4
#define N_EDGES 800000
#define HIDDEN  128
#define N_LAYERS 3

typedef __attribute__((ext_vector_type(16))) _Float16 v16h;
typedef __attribute__((ext_vector_type(4)))  _Float16 v4h;
typedef __attribute__((ext_vector_type(8)))  float    v8f;
typedef __attribute__((ext_vector_type(4)))  float    v4f;

// ---------------- utility kernels ------------------------------------------

__global__ void rgnn_zero4(v4f* __restrict__ p, int n4) {
    int i = blockIdx.x * blockDim.x + threadIdx.x;
    if (i < n4) p[i] = (v4f){0.f, 0.f, 0.f, 0.f};
}

__global__ void rgnn_zero_out(float* __restrict__ p, int n) {
    int i = blockIdx.x * blockDim.x + threadIdx.x;
    if (i < n) p[i] = 0.f;
}

// degree: deg[n*4 + r] = #edges of relation r with dst == n
__global__ void rgnn_degree(const int* __restrict__ edge, float* __restrict__ deg) {
    int r = blockIdx.y;
    int e = blockIdx.x * blockDim.x + threadIdx.x;
    if (e < N_EDGES) {
        int d = edge[(size_t)(r * 2 + 1) * N_EDGES + e];
        atomicAdd(&deg[(size_t)d * N_REL + r], 1.0f);
    }
}

// h16[n][j] = f16( tanh( sum_r deg[n][r] * W_in[r][j] ) )
__global__ void rgnn_input(const float* __restrict__ deg,
                           const float* __restrict__ Win,
                           _Float16* __restrict__ h16) {
    int idx = blockIdx.x * blockDim.x + threadIdx.x;   // N*128 threads
    int n = idx >> 7, j = idx & 127;
    if (n < N_NODES) {
        float s = 0.f;
#pragma unroll
        for (int r = 0; r < N_REL; ++r)
            s += deg[(size_t)n * N_REL + r] * Win[r * HIDDEN + j];
        h16[idx] = (_Float16)tanhf(s);
    }
}

// Pack fp32 128x128 matrices into the CDNA5 16-bit B-operand layout:
// per (ktile, ntile) 32x16 tile, lane L holds column n = L&15,
// halves i=0..15 hold K = ktile*32 + (L>=16 ? 8 : 0) + (i&7) + (i>=8 ? 16 : 0).
// Bp index = ((mat*4 + kt)*8 + nt)*512 + lane*16 + i
__global__ void rgnn_pack(const float* __restrict__ W, _Float16* __restrict__ Bp, int total) {
    int idx = blockIdx.x * blockDim.x + threadIdx.x;
    if (idx < total) {
        int u    = idx & 16383;
        int mat  = idx >> 14;
        int i    = u & 15;
        int lane = (u >> 4) & 31;
        int nt   = (u >> 9) & 7;
        int kt   = (u >> 12) & 3;
        int K = kt * 32 + ((lane >> 4) << 3) + (i & 7) + ((i >> 3) << 4);
        int n = nt * 16 + (lane & 15);
        Bp[idx] = (_Float16)W[(size_t)mat * 16384 + (size_t)K * HIDDEN + n];
    }
}

// comb[i] = (float)h16[i]
__global__ void rgnn_initcomb(const _Float16* __restrict__ h16, float* __restrict__ comb, int n4) {
    int i = blockIdx.x * blockDim.x + threadIdx.x;
    if (i < n4) {
        v4h hv = *(const v4h*)(h16 + (size_t)i * 4);
        v4f f = {(float)hv[0], (float)hv[1], (float)hv[2], (float)hv[3]};
        *((v4f*)comb + i) = f;
    }
}

// segment_sum scatter: one wave per edge, lane c handles 4 features.
__global__ void rgnn_scatter(const int* __restrict__ src, const int* __restrict__ dst,
                             const _Float16* __restrict__ h16, float* __restrict__ agg) {
    int tid = blockIdx.x * blockDim.x + threadIdx.x;
    int e = tid >> 5, c = tid & 31;
    if (e < N_EDGES) {
        int s = src[e], d = dst[e];
        v4h hv = *(const v4h*)(h16 + (size_t)s * HIDDEN + c * 4);
        float* ap = agg + (size_t)d * HIDDEN + c * 4;
        atomicAdd(ap + 0, (float)hv[0]);
        atomicAdd(ap + 1, (float)hv[1]);
        atomicAdd(ap + 2, (float)hv[2]);
        atomicAdd(ap + 3, (float)hv[3]);
    }
}

// ---------------- WMMA GEMM: (16-row tile) x (128x128 packed weights) ------
// A built on the fly from fp32 rows using the 16-bit A-operand striping:
//   lane L: row m = m0 + (L&15), hiK = L>=16
//   halves 0..7  : K = kt*32 +      hiK*8 + 0..7
//   halves 8..15 : K = kt*32 + 16 + hiK*8 + 0..7
// FINAL=false: comb[m][n] += tanh(acc)     (branch GEMM)
// FINAL=true : hout[m][n]  = f16(tanh(acc)) (comb GEMM)
template <bool FINAL>
__global__ __launch_bounds__(256) void rgnn_gemm(const float* __restrict__ A,
                                                 const _Float16* __restrict__ Bp,
                                                 float* __restrict__ comb,
                                                 _Float16* __restrict__ hout,
                                                 int nTiles) {
    int wave = blockIdx.x * (blockDim.x >> 5) + (threadIdx.x >> 5);
    if (wave >= nTiles) return;          // wave-uniform: EXEC stays all-ones
    int lane = threadIdx.x & 31;
    int lrow = lane & 15;
    int hiK  = lane >> 4;
    size_t m0 = (size_t)wave * 16;
    const float* row = A + (m0 + lrow) * HIDDEN;

    v8f acc[8];
#pragma unroll
    for (int nt = 0; nt < 8; ++nt) acc[nt] = (v8f){0.f,0.f,0.f,0.f,0.f,0.f,0.f,0.f};

#pragma unroll
    for (int kt = 0; kt < 4; ++kt) {
        int K0 = kt * 32 + hiK * 8;
        v4f f0 = *(const v4f*)(row + K0);
        v4f f1 = *(const v4f*)(row + K0 + 4);
        v4f f2 = *(const v4f*)(row + K0 + 16);
        v4f f3 = *(const v4f*)(row + K0 + 20);
        v16h a;
#pragma unroll
        for (int i = 0; i < 4; ++i) {
            a[i]      = (_Float16)f0[i];
            a[4 + i]  = (_Float16)f1[i];
            a[8 + i]  = (_Float16)f2[i];
            a[12 + i] = (_Float16)f3[i];
        }
        const _Float16* bbase = Bp + ((size_t)kt * 8 * 32 + lane) * 16;
#pragma unroll
        for (int nt = 0; nt < 8; ++nt) {
            v16h b = *(const v16h*)(bbase + (size_t)nt * 512);
            acc[nt] = __builtin_amdgcn_wmma_f32_16x16x32_f16(
                false, a, false, b, (short)0, acc[nt], false, false);
        }
    }

    // C/D layout: VGPR j -> M = j + 8*hiLane, N = lane&15
    int n0 = lane & 15;
    size_t mbase = m0 + (hiK ? 8 : 0);
#pragma unroll
    for (int nt = 0; nt < 8; ++nt) {
#pragma unroll
        for (int j = 0; j < 8; ++j) {
            size_t off = (mbase + j) * HIDDEN + nt * 16 + n0;
            float t = tanhf(acc[nt][j]);
            if (FINAL) hout[off] = (_Float16)t;
            else       comb[off] += t;
        }
    }
}

// out[j] += partial column sums of h16
__global__ void rgnn_reduce(const _Float16* __restrict__ h16, float* __restrict__ out) {
    int j = threadIdx.x;                 // 128 threads
    float s = 0.f;
    for (int n = blockIdx.x; n < N_NODES; n += gridDim.x)
        s += (float)h16[(size_t)n * HIDDEN + j];
    atomicAdd(&out[j], s);
}

// ---------------------------------------------------------------------------

extern "C" void kernel_launch(void* const* d_in, const int* in_sizes, int n_in,
                              void* d_out, int out_size, void* d_ws, size_t ws_size,
                              hipStream_t stream) {
    const int*   edge     = (const int*)d_in[0];     // (4, 2, 800000) int32
    const float* W_in     = (const float*)d_in[1];   // (4, 128)
    const float* W_branch = (const float*)d_in[2];   // (3, 4, 128, 128)
    const float* W_comb   = (const float*)d_in[3];   // (3, 128, 128)
    float* out = (float*)d_out;                      // (128,)

    // ---- workspace carving (256-byte aligned) ----
    char* ws = (char*)d_ws;
    size_t off = 0;
    auto carve = [&](size_t bytes) -> char* {
        char* p = ws + off;
        off = (off + bytes + 255) & ~(size_t)255;
        return p;
    };
    float*    deg  = (float*)carve((size_t)N_NODES * N_REL * sizeof(float));
    float*    comb = (float*)carve((size_t)N_NODES * HIDDEN * sizeof(float));
    float*    agg  = (float*)carve((size_t)N_NODES * HIDDEN * sizeof(float));
    _Float16* h16  = (_Float16*)carve((size_t)N_NODES * HIDDEN * sizeof(_Float16));
    _Float16* Wb16 = (_Float16*)carve((size_t)N_LAYERS * N_REL * 16384 * sizeof(_Float16));
    _Float16* Wc16 = (_Float16*)carve((size_t)N_LAYERS * 16384 * sizeof(_Float16));

    const int n4_h   = N_NODES * HIDDEN / 4;          // 3,200,000 v4f
    const int n4_deg = N_NODES * N_REL / 4;           // 100,000 v4f
    const int tiles  = N_NODES / 16;                  // 6250
    const int gemmBlocks = (tiles + 7) / 8;           // 8 waves/block

    // 1) degrees
    rgnn_zero4<<<(n4_deg + 255) / 256, 256, 0, stream>>>((v4f*)deg, n4_deg);
    dim3 dgrid(N_EDGES / 256, N_REL);
    rgnn_degree<<<dgrid, 256, 0, stream>>>(edge, deg);

    // 2) pack weights into f16 B-operand layout
    {
        int tb = N_LAYERS * N_REL * 16384;
        rgnn_pack<<<tb / 256, 256, 0, stream>>>(W_branch, Wb16, tb);
        int tc = N_LAYERS * 16384;
        rgnn_pack<<<tc / 256, 256, 0, stream>>>(W_comb, Wc16, tc);
    }

    // 3) input embedding: h = tanh(deg.T @ W_in)
    rgnn_input<<<(N_NODES * HIDDEN) / 256, 256, 0, stream>>>(deg, W_in, h16);

    // 4) layers
    for (int l = 0; l < N_LAYERS; ++l) {
        rgnn_initcomb<<<n4_h / 256, 256, 0, stream>>>(h16, comb, n4_h);
        for (int r = 0; r < N_REL; ++r) {
            const int* src = edge + (size_t)(r * 2 + 0) * N_EDGES;
            const int* dst = edge + (size_t)(r * 2 + 1) * N_EDGES;
            rgnn_zero4<<<n4_h / 256, 256, 0, stream>>>((v4f*)agg, n4_h);
            rgnn_scatter<<<(N_EDGES * 32) / 256, 256, 0, stream>>>(src, dst, h16, agg);
            rgnn_gemm<false><<<gemmBlocks, 256, 0, stream>>>(
                agg, Wb16 + (size_t)(l * N_REL + r) * 16384, comb, (_Float16*)nullptr, tiles);
        }
        rgnn_gemm<true><<<gemmBlocks, 256, 0, stream>>>(
            comb, Wc16 + (size_t)l * 16384, (float*)nullptr, h16, tiles);
    }

    // 5) column-sum reduction
    rgnn_zero_out<<<1, 128, 0, stream>>>(out, HIDDEN);
    rgnn_reduce<<<256, 128, 0, stream>>>(h16, out);
}
